// LPCNet_40235253629017
// MI455X (gfx1250) — matrix operand order repaired and checked by reading, not compiled, not run on hardware
//
#include <hip/hip_runtime.h>
#include <math.h>

// ---------------- constants (LPCNet shapes) ----------------
#define BB   16
#define TT   36
#define FD   20
#define PE   64
#define CD   128
#define TC1  34          // after conv1 (K=3 valid)
#define TC2  32          // after conv2
#define SE   128
#define NS   3
#define UA   384
#define UB   64
#define OUTD 256
#define SS   5120        // 32 * 160
#define ROWS (BB*SS)     // 81920

typedef __attribute__((ext_vector_type(16))) __bf16 v16bf;
typedef __attribute__((ext_vector_type(8)))  float  v8f;

union FragU { uint4 q[2]; v16bf v; };

static __device__ __forceinline__ unsigned short f2bf(float f) {
    union { float f; unsigned u; } c; c.f = f;
    unsigned r = c.u + 0x7FFFu + ((c.u >> 16) & 1u);   // round-to-nearest-even
    return (unsigned short)(r >> 16);
}

// A-matrix fragment (16x32 bf16, MxK). lane&15 = row, lane>>4 = khalf.
// v0..3 hold K = khalf*8 + 0..7 ; v4..7 hold K = 16 + khalf*8 + 0..7.
static __device__ __forceinline__ v16bf load_a_frag(const unsigned short* rowPtr, int kbase, int khalf) {
    FragU u;
    u.q[0] = *reinterpret_cast<const uint4*>(rowPtr + kbase + khalf * 8);
    u.q[1] = *reinterpret_cast<const uint4*>(rowPtr + kbase + 16 + khalf * 8);
    return u.v;
}
// B-matrix fragment (32x16 bf16, KxN). lane&15 = col, lane>>4 = kgroup.
// v0..3 hold K = kgroup*8 + 0..7 ; v4..7 hold K = 16 + kgroup*8 + 0..7.
static __device__ __forceinline__ v16bf load_b_frag(const unsigned short* colPtr, int kbase, int kgroup) {
    FragU u;
    u.q[0] = *reinterpret_cast<const uint4*>(colPtr + kbase + kgroup * 8);
    u.q[1] = *reinterpret_cast<const uint4*>(colPtr + kbase + 16 + kgroup * 8);
    return u.v;
}

static __device__ __forceinline__ float sigmoidf_(float x) { return 1.f / (1.f + expf(-x)); }

// ---------------- fp32 -> bf16 convert ----------------
__global__ void cvt_bf16_kernel(const float* __restrict__ src, unsigned short* __restrict__ dst, int n) {
    int i = blockIdx.x * 256 + threadIdx.x;
    if (i < n) dst[i] = f2bf(src[i]);
}

// ---------------- frame-rate network ----------------
__global__ void conv1_kernel(const float* __restrict__ feat, const int* __restrict__ periods,
                             const float* __restrict__ pemb, const float* __restrict__ w,
                             const float* __restrict__ bias, float* __restrict__ out) {
    int b = blockIdx.x / TC1, t0 = blockIdx.x % TC1, o = threadIdx.x;
    float acc = bias[o];
    for (int k = 0; k < 3; ++k) {
        int t = t0 + k;
        const float* f = feat + (b * TT + t) * FD;
        const float* e = pemb + periods[b * TT + t] * PE;
        for (int i = 0; i < FD; ++i) acc += f[i] * w[(o * (FD + PE) + i) * 3 + k];
        for (int i = 0; i < PE; ++i) acc += e[i] * w[(o * (FD + PE) + FD + i) * 3 + k];
    }
    out[(b * TC1 + t0) * CD + o] = tanhf(acc);
}

__global__ void conv2_kernel(const float* __restrict__ in, const float* __restrict__ w,
                             const float* __restrict__ bias, float* __restrict__ out) {
    int b = blockIdx.x / TC2, t0 = blockIdx.x % TC2, o = threadIdx.x;
    float acc = bias[o];
    for (int k = 0; k < 3; ++k) {
        const float* x = in + (b * TC1 + t0 + k) * CD;
        for (int i = 0; i < CD; ++i) acc += x[i] * w[(o * CD + i) * 3 + k];
    }
    out[(b * TC2 + t0) * CD + o] = tanhf(acc);
}

__global__ void dense2_kernel(const float* __restrict__ in,
                              const float* __restrict__ w1, const float* __restrict__ b1,
                              const float* __restrict__ w2, const float* __restrict__ b2,
                              float* __restrict__ out) {
    __shared__ float v1[CD], v2[CD];
    int idx = blockIdx.x, o = threadIdx.x;
    v1[o] = in[idx * CD + o];
    __syncthreads();
    float a = b1[o];
    for (int i = 0; i < CD; ++i) a += w1[o * CD + i] * v1[i];
    v2[o] = tanhf(a);
    __syncthreads();
    float a2 = b2[o];
    for (int i = 0; i < CD; ++i) a2 += w2[o * CD + i] * v2[i];
    out[idx * CD + o] = tanhf(a2);
}

// ---------------- packing kernels ----------------
__global__ void pack_xa_kernel(const int* __restrict__ signals, const float* __restrict__ semb,
                               const float* __restrict__ c, unsigned short* __restrict__ xa) {
    size_t idx = (size_t)blockIdx.x * 256 + threadIdx.x;   // ROWS*512 total
    int col = (int)(idx & 511);
    size_t row = idx >> 9;
    int b = (int)(row / SS), s = (int)(row % SS);
    float v;
    if (col < 384) {
        int reg = col >> 7, sub = col & 127;
        int sg = signals[row * 3 + reg];
        v = semb[sg * SE + sub];
    } else {
        v = c[((size_t)b * TC2 + (s / 160)) * CD + (col - 384)];
    }
    xa[idx] = f2bf(v);
}

__global__ void pack_xb_kernel(const unsigned short* __restrict__ ya, const float* __restrict__ c,
                               unsigned short* __restrict__ xb) {
    size_t idx = (size_t)blockIdx.x * 256 + threadIdx.x;
    int col = (int)(idx & 511);
    size_t row = idx >> 9;
    if (col < UA) {
        xb[idx] = ya[row * UA + col];
    } else {
        int b = (int)(row / SS), s = (int)(row % SS);
        xb[idx] = f2bf(c[((size_t)b * TC2 + (s / 160)) * CD + (col - UA)]);
    }
}

// ---------------- generic bf16 WMMA GEMM: C[M,N] = A[M,K] * B[N,K]^T + bias ----------------
__global__ void gemm_bf16_kernel(const unsigned short* __restrict__ A, const unsigned short* __restrict__ Bm,
                                 const float* __restrict__ bias, float* __restrict__ C,
                                 int Mt, int Nt, int nkt, int Kd, int N) {
    int wave = threadIdx.x >> 5, lane = threadIdx.x & 31;
    int tile = blockIdx.x * 8 + wave;
    if (tile >= Mt * Nt) return;
    int mt = tile / Nt, nt = tile - mt * Nt;
    int col = lane & 15, hi = lane >> 4, mbase = hi * 8;
    const unsigned short* Arow = A + (size_t)(mt * 16 + col) * Kd;
    const unsigned short* Bcol = Bm + (size_t)(nt * 16 + col) * Kd;
    v8f acc = {};
    for (int kt = 0; kt < nkt; ++kt) {
        v16bf a = load_a_frag(Arow, kt * 32, hi);
        v16bf b = load_b_frag(Bcol, kt * 32, hi);
        acc = __builtin_amdgcn_wmma_f32_16x16x32_bf16(false, a, false, b, (short)0, acc, false, false);
    }
    float bv = bias ? bias[nt * 16 + col] : 0.f;
    for (int r = 0; r < 8; ++r)
        C[(size_t)(mt * 16 + mbase + r) * N + nt * 16 + col] = acc[r] + bv;
}

// ---------------- GRU-A persistent scan: 1 block, 24 waves ----------------
__global__ void __launch_bounds__(768) gru_a_scan_kernel(const float* __restrict__ xp,
                                                         const unsigned short* __restrict__ whh,
                                                         const float* __restrict__ bhh,
                                                         const float* __restrict__ h0,
                                                         unsigned short* __restrict__ y) {
    extern __shared__ char smem[];
    float* hF = (float*)smem;                                       // 16*384 f32
    unsigned short* hH = (unsigned short*)(smem + BB * UA * 4);     // 16*384 bf16
    float* hp = (float*)(smem + BB * UA * 4 + BB * UA * 2);         // 1152*16 f32 ([n][m])
    int tid = threadIdx.x;
    for (int i = tid; i < BB * UA; i += 768) { float v = h0[i]; hF[i] = v; hH[i] = f2bf(v); }
    __syncthreads();
    int wv = tid >> 5, lane = tid & 31, col = lane & 15, hi = lane >> 4, mbase = hi * 8;
    for (int s = 0; s < SS; ++s) {
        // phase 1: hp = h * whh^T via WMMA, 24 waves x 3 tiles = 72 N-tiles
        for (int j = 0; j < 3; ++j) {
            int nt = wv * 3 + j;
            const unsigned short* Bcol = whh + (size_t)(nt * 16 + col) * UA;
            const unsigned short* Arow = hH + col * UA;
            v8f acc = {};
            for (int kt = 0; kt < UA / 32; ++kt) {
                v16bf a = load_a_frag(Arow, kt * 32, hi);
                v16bf b = load_b_frag(Bcol, kt * 32, hi);
                acc = __builtin_amdgcn_wmma_f32_16x16x32_bf16(false, a, false, b, (short)0, acc, false, false);
            }
            for (int r = 0; r < 8; ++r) hp[(nt * 16 + col) * 16 + mbase + r] = acc[r];
        }
        __syncthreads();
        // phase 2: gate update for 16*384 units
        for (int it = 0; it < 8; ++it) {
            int id = tid + it * 768;
            int b = id / UA, u = id - b * UA;
            size_t row = (size_t)b * SS + s;
            const float* xpr = xp + row * (3 * UA);
            float xr = xpr[u], xz = xpr[UA + u], xn = xpr[2 * UA + u];
            float hr = hp[u * 16 + b] + bhh[u];
            float hz = hp[(UA + u) * 16 + b] + bhh[UA + u];
            float hn = hp[(2 * UA + u) * 16 + b] + bhh[2 * UA + u];
            float rg = sigmoidf_(xr + hr);
            float zg = sigmoidf_(xz + hz);
            float ng = tanhf(xn + rg * hn);
            float hnew = (1.f - zg) * ng + zg * hF[b * UA + u];
            hF[b * UA + u] = hnew;
            unsigned short hb = f2bf(hnew);
            hH[b * UA + u] = hb;
            y[row * UA + u] = hb;
        }
        __syncthreads();
    }
}

// ---------------- GRU-B persistent scan: 1 block, 12 waves ----------------
__global__ void __launch_bounds__(384) gru_b_scan_kernel(const float* __restrict__ xp,
                                                         const unsigned short* __restrict__ whh,
                                                         const float* __restrict__ bhh,
                                                         const float* __restrict__ h0,
                                                         unsigned short* __restrict__ y) {
    __shared__ float hF[BB * UB];
    __shared__ __align__(16) unsigned short hH[BB * UB];
    __shared__ float hp[3 * UB * 16];
    int tid = threadIdx.x;
    for (int i = tid; i < BB * UB; i += 384) { float v = h0[i]; hF[i] = v; hH[i] = f2bf(v); }
    __syncthreads();
    int wv = tid >> 5, lane = tid & 31, col = lane & 15, hi = lane >> 4, mbase = hi * 8;
    for (int s = 0; s < SS; ++s) {
        {
            int nt = wv;  // 12 waves, 12 N-tiles (N = 192)
            const unsigned short* Bcol = whh + (size_t)(nt * 16 + col) * UB;
            const unsigned short* Arow = hH + col * UB;
            v8f acc = {};
            for (int kt = 0; kt < UB / 32; ++kt) {
                v16bf a = load_a_frag(Arow, kt * 32, hi);
                v16bf b = load_b_frag(Bcol, kt * 32, hi);
                acc = __builtin_amdgcn_wmma_f32_16x16x32_bf16(false, a, false, b, (short)0, acc, false, false);
            }
            for (int r = 0; r < 8; ++r) hp[(nt * 16 + col) * 16 + mbase + r] = acc[r];
        }
        __syncthreads();
        for (int it = 0; it < 3; ++it) {
            int id = tid + it * 384;
            if (id < BB * UB) {
                int b = id >> 6, u = id & 63;
                size_t row = (size_t)b * SS + s;
                const float* xpr = xp + row * (3 * UB);
                float xr = xpr[u], xz = xpr[UB + u], xn = xpr[2 * UB + u];
                float hr = hp[u * 16 + b] + bhh[u];
                float hz = hp[(UB + u) * 16 + b] + bhh[UB + u];
                float hn = hp[(2 * UB + u) * 16 + b] + bhh[2 * UB + u];
                float rg = sigmoidf_(xr + hr);
                float zg = sigmoidf_(xz + hz);
                float ng = tanhf(xn + rg * hn);
                float hnew = (1.f - zg) * ng + zg * hF[b * UB + u];
                hF[b * UB + u] = hnew;
                unsigned short hb = f2bf(hnew);
                hH[b * UB + u] = hb;
                y[row * UB + u] = hb;
            }
        }
        __syncthreads();
    }
}

// ---------------- fused dual-FC + log-softmax: 1 block per 16 rows ----------------
__global__ void __launch_bounds__(256) dualfc_softmax_kernel(const unsigned short* __restrict__ yb,
                                                             const unsigned short* __restrict__ w1,
                                                             const float* __restrict__ b1,
                                                             const unsigned short* __restrict__ w2,
                                                             const float* __restrict__ b2,
                                                             const float* __restrict__ alpha,
                                                             const float* __restrict__ beta,
                                                             float* __restrict__ out) {
    __shared__ float s1[OUTD * 16];
    __shared__ float s2[OUTD * 16];
    __shared__ float red[256];
    int mt = blockIdx.x, tid = threadIdx.x;
    int wv = tid >> 5, lane = tid & 31, col = lane & 15, hi = lane >> 4, mbase = hi * 8;
    const unsigned short* Arow = yb + (size_t)(mt * 16 + col) * UB;
    for (int j = 0; j < 2; ++j) {
        int nt = wv * 2 + j;    // 8 waves x 2 = 16 N-tiles = 256 cols
        const unsigned short* B1 = w1 + (size_t)(nt * 16 + col) * UB;
        const unsigned short* B2 = w2 + (size_t)(nt * 16 + col) * UB;
        v8f a1 = {}, a2 = {};
        for (int kt = 0; kt < UB / 32; ++kt) {
            v16bf a  = load_a_frag(Arow, kt * 32, hi);
            v16bf f1 = load_b_frag(B1, kt * 32, hi);
            v16bf f2 = load_b_frag(B2, kt * 32, hi);
            a1 = __builtin_amdgcn_wmma_f32_16x16x32_bf16(false, a, false, f1, (short)0, a1, false, false);
            a2 = __builtin_amdgcn_wmma_f32_16x16x32_bf16(false, a, false, f2, (short)0, a2, false, false);
        }
        for (int r = 0; r < 8; ++r) {
            s1[(nt * 16 + col) * 16 + mbase + r] = a1[r];
            s2[(nt * 16 + col) * 16 + mbase + r] = a2[r];
        }
    }
    __syncthreads();
    int row = tid >> 4, sub = tid & 15;
    float g[16], lmax = -1e30f;
    for (int cc = 0; cc < 16; ++cc) {
        int c = sub * 16 + cc;
        float v = alpha[c] * tanhf(s1[c * 16 + row] + b1[c]) + beta[c] * tanhf(s2[c * 16 + row] + b2[c]);
        g[cc] = v; lmax = fmaxf(lmax, v);
    }
    red[tid] = lmax;
    __syncthreads();
    for (int off = 8; off >= 1; off >>= 1) {
        if (sub < off) red[tid] = fmaxf(red[tid], red[tid + off]);
        __syncthreads();
    }
    float m = red[row * 16];
    __syncthreads();
    float ls = 0.f;
    for (int cc = 0; cc < 16; ++cc) ls += expf(g[cc] - m);
    red[tid] = ls;
    __syncthreads();
    for (int off = 8; off >= 1; off >>= 1) {
        if (sub < off) red[tid] += red[tid + off];
        __syncthreads();
    }
    float lse = logf(red[row * 16]);
    size_t base = (size_t)(mt * 16 + row) * OUTD;
    for (int cc = 0; cc < 16; ++cc) out[base + sub * 16 + cc] = g[cc] - m - lse;
}

// ---------------- host launch ----------------
extern "C" void kernel_launch(void* const* d_in, const int* in_sizes, int n_in,
                              void* d_out, int out_size, void* d_ws, size_t ws_size,
                              hipStream_t stream) {
    const float* features   = (const float*)d_in[0];
    const int*   periods    = (const int*)d_in[1];
    const int*   signals    = (const int*)d_in[2];
    const float* h0a        = (const float*)d_in[3];
    const float* h0b        = (const float*)d_in[4];
    const float* period_emb = (const float*)d_in[5];
    const float* signal_emb = (const float*)d_in[6];
    const float* conv1_w    = (const float*)d_in[7];
    const float* conv1_b    = (const float*)d_in[8];
    const float* conv2_w    = (const float*)d_in[9];
    const float* conv2_b    = (const float*)d_in[10];
    const float* fd1_w      = (const float*)d_in[11];
    const float* fd1_b      = (const float*)d_in[12];
    const float* fd2_w      = (const float*)d_in[13];
    const float* fd2_b      = (const float*)d_in[14];
    const float* grua_wih   = (const float*)d_in[15];
    const float* grua_whh   = (const float*)d_in[16];
    const float* grua_bih   = (const float*)d_in[17];
    const float* grua_bhh   = (const float*)d_in[18];
    const float* grub_wih   = (const float*)d_in[19];
    const float* grub_whh   = (const float*)d_in[20];
    const float* grub_bih   = (const float*)d_in[21];
    const float* grub_bhh   = (const float*)d_in[22];
    const float* fc1_w      = (const float*)d_in[23];
    const float* fc1_b      = (const float*)d_in[24];
    const float* fc2_w      = (const float*)d_in[25];
    const float* fc2_b      = (const float*)d_in[26];
    const float* alpha      = (const float*)d_in[27];
    const float* beta       = (const float*)d_in[28];
    float* out = (float*)d_out;

    char* base = (char*)d_ws;
    size_t off = 0;
    auto carve = [&](size_t bytes) -> char* {
        char* p = base + off;
        off = (off + bytes + 255) & ~(size_t)255;
        return p;
    };
    float*          tmp1    = (float*)carve((size_t)BB * TC1 * CD * 4);
    float*          tmp2    = (float*)carve((size_t)BB * TC2 * CD * 4);
    float*          cF      = (float*)carve((size_t)BB * TC2 * CD * 4);
    unsigned short* wihA_bf = (unsigned short*)carve((size_t)3 * UA * 512 * 2);
    unsigned short* whhA_bf = (unsigned short*)carve((size_t)3 * UA * UA * 2);
    unsigned short* wihB_bf = (unsigned short*)carve((size_t)3 * UB * 512 * 2);
    unsigned short* whhB_bf = (unsigned short*)carve((size_t)3 * UB * UB * 2);
    unsigned short* fc1_bf  = (unsigned short*)carve((size_t)OUTD * UB * 2);
    unsigned short* fc2_bf  = (unsigned short*)carve((size_t)OUTD * UB * 2);
    unsigned short* xa      = (unsigned short*)carve((size_t)ROWS * 512 * 2);
    float*          xpa     = (float*)carve((size_t)ROWS * 3 * UA * 4);
    unsigned short* ya      = (unsigned short*)carve((size_t)ROWS * UA * 2);
    unsigned short* xb      = (unsigned short*)carve((size_t)ROWS * 512 * 2);
    float*          xpb     = (float*)carve((size_t)ROWS * 3 * UB * 4);
    unsigned short* yb      = (unsigned short*)carve((size_t)ROWS * UB * 2);

    auto cvt = [&](const float* s, unsigned short* d, int n) {
        cvt_bf16_kernel<<<(n + 255) / 256, 256, 0, stream>>>(s, d, n);
    };
    cvt(grua_wih, wihA_bf, 3 * UA * 512);
    cvt(grua_whh, whhA_bf, 3 * UA * UA);
    cvt(grub_wih, wihB_bf, 3 * UB * 512);
    cvt(grub_whh, whhB_bf, 3 * UB * UB);
    cvt(fc1_w, fc1_bf, OUTD * UB);
    cvt(fc2_w, fc2_bf, OUTD * UB);

    // frame-rate network
    conv1_kernel<<<BB * TC1, CD, 0, stream>>>(features, periods, period_emb, conv1_w, conv1_b, tmp1);
    conv2_kernel<<<BB * TC2, CD, 0, stream>>>(tmp1, conv2_w, conv2_b, tmp2);
    dense2_kernel<<<BB * TC2, CD, 0, stream>>>(tmp2, fd1_w, fd1_b, fd2_w, fd2_b, cF);

    // GRU-A: pack input, project (WMMA), scan (WMMA recurrence)
    pack_xa_kernel<<<(unsigned)((size_t)ROWS * 512 / 256), 256, 0, stream>>>(signals, signal_emb, cF, xa);
    {
        int Mt = ROWS / 16, Nt = (3 * UA) / 16, tiles = Mt * Nt;
        gemm_bf16_kernel<<<(tiles + 7) / 8, 256, 0, stream>>>(xa, wihA_bf, grua_bih, xpa, Mt, Nt, 512 / 32, 512, 3 * UA);
    }
    {
        size_t lds = (size_t)BB * UA * 4 + (size_t)BB * UA * 2 + (size_t)3 * UA * 16 * 4;  // ~110 KB
        gru_a_scan_kernel<<<1, 768, lds, stream>>>(xpa, whhA_bf, grua_bhh, h0a, ya);
    }

    // GRU-B
    pack_xb_kernel<<<(unsigned)((size_t)ROWS * 512 / 256), 256, 0, stream>>>(ya, cF, xb);
    {
        int Mt = ROWS / 16, Nt = (3 * UB) / 16, tiles = Mt * Nt;
        gemm_bf16_kernel<<<(tiles + 7) / 8, 256, 0, stream>>>(xb, wihB_bf, grub_bih, xpb, Mt, Nt, 512 / 32, 512, 3 * UB);
    }
    gru_b_scan_kernel<<<1, 384, 0, stream>>>(xpb, whhB_bf, grub_bhh, h0b, yb);

    // dual FC + log-softmax
    dualfc_softmax_kernel<<<ROWS / 16, 256, 0, stream>>>(yb, fc1_bf, fc1_b, fc2_bf, fc2_b, alpha, beta, out);

    (void)in_sizes; (void)n_in; (void)out_size; (void)ws_size;
}
